// GridPool_2508260901457
// MI455X (gfx1250) — compile-verified, matrix operand order checked by reading, not run on hardware
//
#include <hip/hip_runtime.h>
#include <hip/hip_bf16.h>

typedef __attribute__((ext_vector_type(16))) _Float16 v16h;
typedef __attribute__((ext_vector_type(8)))  float    v8f;

#define MAXS  (1 << 18)          // dense voxel-key space upper bound (actual S ~ 74K)
#define NBLK  (MAXS / 256)
#define EPSV  1e-5f
#define GEMM_BLOCKS 2048

// ---------------- helpers ----------------
__device__ __forceinline__ int batch_of(int i, const int* offset, int B) {
    int b = 0;
    while (b < B - 1 && i >= offset[b]) b++;
    return b;
}

// Build the per-lane A fragment for a 16-row tile (zero rows >= N: contributes 0 to stats/max)
__device__ __forceinline__ v16h load_a_frag(const float* __restrict__ feat,
                                            int tile, int lane, int kb, int N) {
    int row = tile * 16 + (lane & 15);
    v16h a;
    if (row < N) {
        const float* fr = feat + (size_t)row * 32;
#pragma unroll
        for (int e = 0; e < 8; e++) a[e]     = (_Float16)fr[kb + e];
#pragma unroll
        for (int e = 0; e < 8; e++) a[8 + e] = (_Float16)fr[16 + kb + e];
    } else {
#pragma unroll
        for (int e = 0; e < 16; e++) a[e] = (_Float16)0.f;
    }
    return a;
}

// ---------------- 0: init workspace / outputs ----------------
__global__ __launch_bounds__(256)
void gp_init(float* sum, float* sq, unsigned* bmin, int* vmax, int* bincnt,
             int* occ, unsigned* cnt, int* cbatch, float* csum, int* bsums,
             float* out_feat, long long M) {
    long long g  = (long long)blockIdx.x * 256 + threadIdx.x;
    long long gs = (long long)gridDim.x * 256;
    for (long long i = g; i < MAXS; i += gs) { occ[i] = 0; cnt[i] = 0u; cbatch[i] = 0x7fffffff; }
    for (long long i = g; i < 3LL * MAXS; i += gs) csum[i] = 0.f;
    for (long long i = g; i < M * 64;   i += gs) out_feat[i] = 0.f;   // lower bound for ReLU>=0 max
    for (long long i = g; i < NBLK;     i += gs) bsums[i] = 0;
    if (g < 64)  { sum[g] = 0.f; sq[g] = 0.f; bincnt[g] = 0; }
    if (g < 192) bmin[g] = 0x7f800000u;                               // +inf bits (coords >= 0)
    if (g < 3)   vmax[g] = 0;
}

// ---------------- 1: persistent WMMA GEMM pass 1 -> per-channel sum / sumsq ----------------
__global__ __launch_bounds__(128)
void gp_gemm_stats(const float* __restrict__ feat, const float* __restrict__ W,
                   float* sum, float* sq, int N, int T) {
    __shared__ _Float16 wl[32 * 64];
    __shared__ float s_sum[64], s_sq[64];
    int tid = threadIdx.x;
    if (tid < 64) { s_sum[tid] = 0.f; s_sq[tid] = 0.f; }
    for (int i = tid; i < 32 * 64; i += 128) wl[i] = (_Float16)W[i];
    __syncthreads();

    int wave = tid >> 5, lane = tid & 31;
    int kb   = (lane < 16) ? 0 : 8;              // 16-bit operand layout: K halves by lane group
    int ncol = lane & 15;

    // B fragments built ONCE per wave, resident in registers across all row tiles
    v16h bf[4];
#pragma unroll
    for (int t = 0; t < 4; t++) {
        int col = t * 16 + ncol;
#pragma unroll
        for (int e = 0; e < 8; e++) bf[t][e]     = wl[(kb + e) * 64 + col];
#pragma unroll
        for (int e = 0; e < 8; e++) bf[t][8 + e] = wl[(16 + kb + e) * 64 + col];
    }

    float lsum[4] = {0.f, 0.f, 0.f, 0.f};
    float lsq[4]  = {0.f, 0.f, 0.f, 0.f};
    int nw = gridDim.x * 4;
    for (int tile = blockIdx.x * 4 + wave; tile < T; tile += nw) {
        v16h a = load_a_frag(feat, tile, lane, kb, N);
#pragma unroll
        for (int t = 0; t < 4; t++) {
            v8f acc = {};
            acc = __builtin_amdgcn_wmma_f32_16x16x32_f16(false, a, false, bf[t],
                                                         (short)0, acc, false, false);
            float s = 0.f, q = 0.f;
#pragma unroll
            for (int r = 0; r < 8; r++) { float d = acc[r]; s += d; q += d * d; }
            lsum[t] += s;
            lsq[t]  += q;
        }
    }

    // one reduction per wave at kernel end: combine M=0..7 / M=8..15 halves, then LDS, then global
#pragma unroll
    for (int t = 0; t < 4; t++) {
        float s = lsum[t] + __shfl_xor(lsum[t], 16, 32);
        float q = lsq[t]  + __shfl_xor(lsq[t],  16, 32);
        if (lane < 16) {
            atomicAdd(&s_sum[t * 16 + ncol], s);
            atomicAdd(&s_sq[t * 16 + ncol],  q);
        }
    }
    __syncthreads();
    if (tid < 64) { atomicAdd(&sum[tid], s_sum[tid]); atomicAdd(&sq[tid], s_sq[tid]); }
}

// ---------------- 2: finalize BN scale/shift ----------------
__global__ void gp_bn_finalize(const float* sum, const float* sq,
                               const float* gamma, const float* beta,
                               float* scale, float* shift, int N) {
    int c = threadIdx.x;
    if (c >= 64) return;
    float invN = 1.f / (float)N;
    float mean = sum[c] * invN;
    float var  = sq[c] * invN - mean * mean;
    float s = gamma[c] * rsqrtf(var + EPSV);
    scale[c] = s;
    shift[c] = beta[c] - mean * s;
}

// ---------------- 3: per-batch min coord ----------------
__global__ __launch_bounds__(256)
void gp_batch_min(const float* coord, const int* offset, unsigned* bmin, int N, int B) {
    int i = blockIdx.x * 256 + threadIdx.x;
    if (i >= N) return;
    int b = batch_of(i, offset, B);
#pragma unroll
    for (int d = 0; d < 3; d++)
        atomicMin(&bmin[b * 3 + d], __float_as_uint(coord[(size_t)i * 3 + d]));
}

// ---------------- 4: global voxel-extent max ----------------
__global__ __launch_bounds__(256)
void gp_vmax(const float* coord, const int* offset, const unsigned* bmin,
             const float* gsize, int* vmax, int N, int B) {
    int i = blockIdx.x * 256 + threadIdx.x;
    if (i >= N) return;
    int b = batch_of(i, offset, B);
    float gs = gsize[0];
#pragma unroll
    for (int d = 0; d < 3; d++) {
        float st = __uint_as_float(bmin[b * 3 + d]);
        int v = (int)floorf((coord[(size_t)i * 3 + d] - st) / gs);
        atomicMax(&vmax[d], v);
    }
}

// ---------------- 5: grid dims + key-space size ----------------
__global__ void gp_dims(const int* vmax, int* meta, int B) {
    if (threadIdx.x != 0 || blockIdx.x != 0) return;
    int D0 = vmax[0] + 1, D1 = vmax[1] + 1, D2 = vmax[2] + 1;
    long long S = (long long)B * D0 * D1 * D2;
    if (S > MAXS) S = MAXS;
    meta[0] = D0; meta[1] = D1; meta[2] = D2; meta[3] = (int)S;
}

// ---------------- 6: voxel keys + occupancy ----------------
__global__ __launch_bounds__(256)
void gp_keys(const float* coord, const int* offset, const unsigned* bmin,
             const float* gsize, const int* meta, int* vkey, int* occ, int N, int B) {
    int i = blockIdx.x * 256 + threadIdx.x;
    if (i >= N) return;
    int b = batch_of(i, offset, B);
    float gs = gsize[0];
    int v[3];
#pragma unroll
    for (int d = 0; d < 3; d++) {
        float st = __uint_as_float(bmin[b * 3 + d]);
        v[d] = (int)floorf((coord[(size_t)i * 3 + d] - st) / gs);
    }
    int D0 = meta[0], D1 = meta[1], D2 = meta[2], S = meta[3];
    long long key = (((long long)b * D0 + v[0]) * D1 + v[1]) * D2 + v[2];
    int k = (key >= 0 && key < S) ? (int)key : (S - 1);
    vkey[i] = k;
    occ[k] = 1;
}

// ---------------- 7: per-block exclusive scan of occupancy ----------------
__global__ __launch_bounds__(256)
void gp_scan_blocks(const int* occ, int* rank, int* bsums, const int* meta) {
    __shared__ int sh[256];
    int S = meta[3];
    int gid = blockIdx.x * 256 + threadIdx.x;
    int x = (gid < S) ? occ[gid] : 0;
    sh[threadIdx.x] = x;
    __syncthreads();
    for (int off = 1; off < 256; off <<= 1) {
        int t = (threadIdx.x >= off) ? sh[threadIdx.x - off] : 0;
        __syncthreads();
        sh[threadIdx.x] += t;
        __syncthreads();
    }
    int incl = sh[threadIdx.x];
    rank[gid] = incl - x;
    if (threadIdx.x == 255) bsums[blockIdx.x] = incl;
}

// ---------------- 8: scan of block sums (single block) ----------------
__global__ __launch_bounds__(256)
void gp_scan_bsums(int* bsums, int* meta) {
    __shared__ int sh[256];
    __shared__ int carry;
    if (threadIdx.x == 0) carry = 0;
    __syncthreads();
    for (int base = 0; base < NBLK; base += 256) {
        int x = bsums[base + threadIdx.x];
        sh[threadIdx.x] = x;
        __syncthreads();
        for (int off = 1; off < 256; off <<= 1) {
            int t = (threadIdx.x >= off) ? sh[threadIdx.x - off] : 0;
            __syncthreads();
            sh[threadIdx.x] += t;
            __syncthreads();
        }
        int incl = sh[threadIdx.x];
        bsums[base + threadIdx.x] = incl - x + carry;
        __syncthreads();
        if (threadIdx.x == 255) carry += incl;
        __syncthreads();
    }
    if (threadIdx.x == 0) meta[4] = carry;   // device-computed M
}

// ---------------- 9: add block offsets ----------------
__global__ __launch_bounds__(256)
void gp_scan_add(int* rank, const int* bsums) {
    int gid = blockIdx.x * 256 + threadIdx.x;
    rank[gid] += bsums[blockIdx.x];
}

// ---------------- 10: inverse + coord/count/batch aggregation ----------------
__global__ __launch_bounds__(256)
void gp_point_agg(const float* coord, const int* offset, const int* vkey,
                  const int* rank, int* inv, float* out_inv,
                  float* csum, unsigned* cnt, int* cbatch, int N, int B) {
    int i = blockIdx.x * 256 + threadIdx.x;
    if (i >= N) return;
    int c = rank[vkey[i]];
    inv[i] = c;
    out_inv[i] = (float)c;
#pragma unroll
    for (int d = 0; d < 3; d++)
        atomicAdd(&csum[(size_t)c * 3 + d], coord[(size_t)i * 3 + d]);
    atomicAdd(&cnt[c], 1u);
    atomicMin(&cbatch[c], batch_of(i, offset, B));
}

// ---------------- 11: persistent WMMA GEMM pass 2 -> BN/ReLU -> segment max ----------------
__global__ __launch_bounds__(128)
void gp_gemm_bn_max(const float* __restrict__ feat, const float* __restrict__ W,
                    const float* scale, const float* shift, const int* inv,
                    float* out_feat, int N, int T) {
    __shared__ _Float16 wl[32 * 64];
    int tid = threadIdx.x;
    for (int i = tid; i < 32 * 64; i += 128) wl[i] = (_Float16)W[i];
    __syncthreads();

    int wave = tid >> 5, lane = tid & 31;
    int kb   = (lane < 16) ? 0 : 8;
    int ncol = lane & 15;

    v16h bf[4];
    float sc[4], sf[4];
#pragma unroll
    for (int t = 0; t < 4; t++) {
        int col = t * 16 + ncol;
#pragma unroll
        for (int e = 0; e < 8; e++) bf[t][e]     = wl[(kb + e) * 64 + col];
#pragma unroll
        for (int e = 0; e < 8; e++) bf[t][8 + e] = wl[(16 + kb + e) * 64 + col];
        sc[t] = scale[col];
        sf[t] = shift[col];
    }

    int nw = gridDim.x * 4;
    for (int tile = blockIdx.x * 4 + wave; tile < T; tile += nw) {
        v16h a = load_a_frag(feat, tile, lane, kb, N);
#pragma unroll
        for (int t = 0; t < 4; t++) {
            int col = t * 16 + ncol;
            v8f acc = {};
            acc = __builtin_amdgcn_wmma_f32_16x16x32_f16(false, a, false, bf[t],
                                                         (short)0, acc, false, false);
#pragma unroll
            for (int r = 0; r < 8; r++) {
                int m = (lane < 16) ? r : (8 + r);
                int row = tile * 16 + m;
                if (row < N) {
                    float y = acc[r] * sc[t] + sf[t];
                    y = y > 0.f ? y : 0.f;
                    int c = inv[row];
                    // ReLU output >= 0: uint bit-pattern max is monotone, init 0
                    atomicMax((unsigned int*)&out_feat[(size_t)c * 64 + col],
                              __float_as_uint(y));
                }
            }
        }
    }
}

// ---------------- 12: bincount of cluster batches ----------------
__global__ __launch_bounds__(256)
void gp_bincount(const int* cbatch, int* bincnt, int M) {
    int c = blockIdx.x * 256 + threadIdx.x;
    if (c >= M) return;
    int b = cbatch[c];
    if (b >= 0 && b < 64) atomicAdd(&bincnt[b], 1);
}

// ---------------- 13: offset_out cumsum ----------------
__global__ void gp_offset_out(const int* bincnt, float* out_off, int B) {
    if (threadIdx.x != 0 || blockIdx.x != 0) return;
    int acc = 0;
    for (int b = 0; b < B; b++) { acc += bincnt[b]; out_off[b] = (float)acc; }
}

// ---------------- 14: coord_out = mean ----------------
__global__ __launch_bounds__(256)
void gp_coord_out(const float* csum, const unsigned* cnt, float* out_coord, int M) {
    int c = blockIdx.x * 256 + threadIdx.x;
    if (c >= M) return;
    unsigned n = cnt[c];
    float invn = n ? 1.f / (float)n : 0.f;
#pragma unroll
    for (int d = 0; d < 3; d++)
        out_coord[(size_t)c * 3 + d] = csum[(size_t)c * 3 + d] * invn;
}

extern "C" void kernel_launch(void* const* d_in, const int* in_sizes, int n_in,
                              void* d_out, int out_size, void* d_ws, size_t ws_size,
                              hipStream_t stream) {
    const float* coord  = (const float*)d_in[0];
    const float* feat   = (const float*)d_in[1];
    const int*   offset = (const int*)  d_in[2];
    const float* W      = (const float*)d_in[3];
    const float* gamma  = (const float*)d_in[4];
    const float* beta   = (const float*)d_in[5];
    const float* gsize  = (const float*)d_in[6];

    int N = in_sizes[0] / 3;
    int B = in_sizes[2];
    int M = (out_size - B - N) / 67;         // 3M + 64M + B + N == out_size
    if (M < 1) M = 1;

    float* out       = (float*)d_out;
    float* out_coord = out;
    float* out_feat  = out + (size_t)3 * M;
    float* out_off   = out + (size_t)67 * M;
    float* out_inv   = out_off + B;

    // workspace layout (element offsets, 4B each)
    float*    fb = (float*)d_ws;
    int*      ib = (int*)d_ws;
    unsigned* ub = (unsigned*)d_ws;
    size_t oSUM = 0, oSQ = 64, oSCALE = 128, oSHIFT = 192;
    size_t oBMIN = 256, oVMAX = 512, oMETA = 520, oBIN = 576;
    size_t oBS   = 1024;
    size_t oKEY  = oBS + NBLK;
    size_t oINV  = oKEY + (size_t)N;
    size_t oOCC  = oINV + (size_t)N;
    size_t oRANK = oOCC + MAXS;
    size_t oCNT  = oRANK + MAXS;
    size_t oCB   = oCNT + MAXS;
    size_t oCS   = oCB + MAXS;

    float*    w_sum    = fb + oSUM;   float* w_sq    = fb + oSQ;
    float*    w_scale  = fb + oSCALE; float* w_shift = fb + oSHIFT;
    unsigned* w_bmin   = ub + oBMIN;  int*   w_vmax  = ib + oVMAX;
    int*      w_meta   = ib + oMETA;  int*   w_bin   = ib + oBIN;
    int*      w_bsums  = ib + oBS;    int*   w_key   = ib + oKEY;
    int*      w_inv    = ib + oINV;   int*   w_occ   = ib + oOCC;
    int*      w_rank   = ib + oRANK;  unsigned* w_cnt = ub + oCNT;
    int*      w_cb     = ib + oCB;    float* w_csum  = fb + oCS;

    int nb256 = (N + 255) / 256;
    int mb256 = (M + 255) / 256;
    int T     = (N + 15) / 16;                 // 16-row WMMA tiles
    int gb    = GEMM_BLOCKS;
    if (gb > (T + 3) / 4) gb = (T + 3) / 4;    // don't launch idle blocks

    gp_init<<<2048, 256, 0, stream>>>(w_sum, w_sq, w_bmin, w_vmax, w_bin,
                                      w_occ, w_cnt, w_cb, w_csum, w_bsums,
                                      out_feat, (long long)M);
    gp_gemm_stats<<<gb, 128, 0, stream>>>(feat, W, w_sum, w_sq, N, T);
    gp_bn_finalize<<<1, 64, 0, stream>>>(w_sum, w_sq, gamma, beta, w_scale, w_shift, N);
    gp_batch_min<<<nb256, 256, 0, stream>>>(coord, offset, w_bmin, N, B);
    gp_vmax<<<nb256, 256, 0, stream>>>(coord, offset, w_bmin, gsize, w_vmax, N, B);
    gp_dims<<<1, 1, 0, stream>>>(w_vmax, w_meta, B);
    gp_keys<<<nb256, 256, 0, stream>>>(coord, offset, w_bmin, gsize, w_meta, w_key, w_occ, N, B);
    gp_scan_blocks<<<NBLK, 256, 0, stream>>>(w_occ, w_rank, w_bsums, w_meta);
    gp_scan_bsums<<<1, 256, 0, stream>>>(w_bsums, w_meta);
    gp_scan_add<<<NBLK, 256, 0, stream>>>(w_rank, w_bsums);
    gp_point_agg<<<nb256, 256, 0, stream>>>(coord, offset, w_key, w_rank, w_inv, out_inv,
                                            w_csum, w_cnt, w_cb, N, B);
    gp_gemm_bn_max<<<gb, 128, 0, stream>>>(feat, W, w_scale, w_shift, w_inv, out_feat, N, T);
    gp_bincount<<<mb256, 256, 0, stream>>>(w_cb, w_bin, M);
    gp_offset_out<<<1, 1, 0, stream>>>(w_bin, out_off, B);
    gp_coord_out<<<mb256, 256, 0, stream>>>(w_csum, w_cnt, out_coord, M);
}
